// SelfAttention_49014166782592
// MI455X (gfx1250) — matrix operand order repaired and checked by reading, not compiled, not run on hardware
//
#include <hip/hip_runtime.h>
#include <hip/hip_bf16.h>

#define BATCH 4
#define SEQ   2048
#define EMBED 1024
#define HEADS 16
#define HD    64   // head dim

typedef __attribute__((ext_vector_type(16))) _Float16 v16h;
typedef __attribute__((ext_vector_type(8)))  _Float16 v8h;
typedef __attribute__((ext_vector_type(8)))  float    v8f;
typedef int vint4 __attribute__((vector_size(16)));   // matches builtin param

static __device__ __forceinline__ v8f wmma_f16(v16h a, v16h b, v8f c) {
  // D = A(16x32 f16) * B(32x16 f16) + C(16x16 f32)
  return __builtin_amdgcn_wmma_f32_16x16x32_f16(false, a, false, b, (short)0, c,
                                                false, false);
}

// A fragment (16x32, f16, row-major source). Caller passes the pointer to the
// lane's row; lanes 0-15 take K = {0..15}+kbase, lanes 16-31 take
// K = {16..31}+kbase  (contiguous, matching the B fragments).
static __device__ __forceinline__ v16h load_a_f16(const _Float16* row_ptr,
                                                  int kbase, int hi) {
  const _Float16* p0 = row_ptr + kbase + hi * 16;
  v8h c0 = *(const v8h*)p0;
  v8h c1 = *(const v8h*)(p0 + 8);
  v16h a;
#pragma unroll
  for (int i = 0; i < 8; ++i) { a[i] = c0[i]; a[8 + i] = c1[i]; }
  return a;
}

// 16-byte global -> LDS copy. Prefer the CDNA5 async-DMA path (ASYNCcnt
// tracked, no VGPR staging); fall back to a synchronous copy otherwise.
static __device__ __forceinline__ void copy16_g2l(const _Float16* g, _Float16* l) {
#if __has_builtin(__builtin_amdgcn_global_load_async_to_lds_b128)
  __builtin_amdgcn_global_load_async_to_lds_b128(
      (__attribute__((address_space(1))) vint4*)(g),
      (__attribute__((address_space(3))) vint4*)(l), 0, 0);
#else
  *(uint4*)l = *(const uint4*)g;
#endif
}

static __device__ __forceinline__ void wait_async_all() {
#if __has_builtin(__builtin_amdgcn_s_wait_asynccnt)
  __builtin_amdgcn_s_wait_asynccnt(0);
#elif defined(__gfx1250__)
  asm volatile("s_wait_asynccnt 0" ::: "memory");
#endif
}

// -------------------------------------------------------------------------
// f32 -> f16 weight conversion
__global__ void cvt_f32_to_f16(const float* __restrict__ src,
                               _Float16* __restrict__ dst, int n) {
  int i = blockIdx.x * blockDim.x + threadIdx.x;
  if (i < n) dst[i] = (_Float16)src[i];
}

// -------------------------------------------------------------------------
// Per-head projection: out[m, e] = sum_d X[m, d] * W[e, d], m = ((b*S+l)*H+h).
// One wave computes a 16x64 tile. Output scattered to [B,H,S,D] f16
// (vtranspose=0) or [B,H,D,S] f16 (vtranspose=1, for V).
__global__ __launch_bounds__(128) void proj16(const float* __restrict__ X,
                                              const _Float16* __restrict__ W16,
                                              _Float16* __restrict__ out,
                                              int vtranspose) {
  const int tid  = threadIdx.x;
  const int wave = tid >> 5, lane = tid & 31;
  const int ln = lane & 15, hi = lane >> 4;
  const int m0 = (blockIdx.x * 4 + wave) * 16;

  // A fragments: 16 rows of X, f32 -> f16 on the fly
  v16h a[2];
  {
    const float* rp = X + (size_t)(m0 + ln) * HD;
#pragma unroll
    for (int ks = 0; ks < 2; ++ks) {
      const float* p0 = rp + ks * 32 + hi * 16;
      v8f f0 = *(const v8f*)p0;
      v8f f1 = *(const v8f*)(p0 + 8);
      v16h t;
#pragma unroll
      for (int i = 0; i < 8; ++i) {
        t[i] = (_Float16)f0[i];
        t[8 + i] = (_Float16)f1[i];
      }
      a[ks] = t;
    }
  }

  v8f acc[4];
#pragma unroll
  for (int nt = 0; nt < 4; ++nt) {
    v8f c = {};
#pragma unroll
    for (int ks = 0; ks < 2; ++ks) {
      // B[k=d, n=e]: lane n = nt*16+ln -> W row e, contiguous 16 f16
      v16h bf = *(const v16h*)(W16 + (size_t)(nt * 16 + ln) * HD + ks * 32 + hi * 16);
      c = wmma_f16(a[ks], bf, c);
    }
    acc[nt] = c;
  }

#pragma unroll
  for (int nt = 0; nt < 4; ++nt) {
    const int d = nt * 16 + ln;
#pragma unroll
    for (int i = 0; i < 8; ++i) {
      const int m = m0 + i + hi * 8;
      const int h = m & (HEADS - 1);
      const int t = m >> 4;          // b*S + l
      const int l = t & (SEQ - 1);
      const int b = t >> 11;         // SEQ = 2048
      size_t idx;
      if (vtranspose)
        idx = ((size_t)(b * HEADS + h) * HD + d) * SEQ + l;
      else
        idx = ((size_t)(b * HEADS + h) * SEQ + l) * HD + d;
      out[idx] = (_Float16)acc[nt][i];
    }
  }
}

// -------------------------------------------------------------------------
// Flash attention: grid (S/64, H, B); 4 waves x 16 q-rows; 64-key KV tiles
// double-buffered in LDS with async global->LDS DMA so tile t+1's fetch hides
// behind tile t's WMMAs. Online softmax in registers via half-wave shuffles.
__global__ __launch_bounds__(128) void flash_attn(const _Float16* __restrict__ Qh,
                                                  const _Float16* __restrict__ Kh,
                                                  const _Float16* __restrict__ Vt,
                                                  const int* __restrict__ mask,
                                                  _Float16* __restrict__ ctx) {
  __shared__ __align__(32) _Float16 sK[2][64 * 64];   // [kv][d]
  __shared__ __align__(32) _Float16 sV[2][64 * 64];   // [d][kv]  (V transposed)
  __shared__ __align__(32) _Float16 sP[4][16 * 64];   // per-wave P staging

  const int tid  = threadIdx.x;
  const int wave = tid >> 5, lane = tid & 31;
  const int ln = lane & 15, hi = lane >> 4;
  const int h = blockIdx.y, b = blockIdx.z;
  const int q0 = blockIdx.x * 64;
  const size_t bh = (size_t)(b * HEADS + h);
  const _Float16* Qb = Qh + bh * SEQ * HD;
  const _Float16* Kb = Kh + bh * SEQ * HD;
  const _Float16* Vb = Vt + bh * HD * SEQ;
  const int* mb = mask + (size_t)b * SEQ * SEQ;

  // issue the async copies for one 64-key tile into buffer `buf`
  auto issue_tile = [&](int kv0, int buf) {
#pragma unroll
    for (int it = 0; it < 4; ++it) {
      const int ch = it * 128 + tid;           // 0..511 16B-chunk id
      const int r = ch >> 3, c8 = (ch & 7) * 8;
      copy16_g2l(&Kb[(size_t)(kv0 + r) * HD + c8], &sK[buf][r * 64 + c8]);
      copy16_g2l(&Vb[(size_t)r * SEQ + kv0 + c8], &sV[buf][r * 64 + c8]);
    }
  };

  // Q fragments (16 rows x 64 d), reused across all KV tiles
  v16h qa[2];
  {
    const _Float16* rp = Qb + (size_t)(q0 + wave * 16 + ln) * HD;
#pragma unroll
    for (int ks = 0; ks < 2; ++ks) qa[ks] = load_a_f16(rp, ks * 32, hi);
  }

  v8f o[4] = {};
  float rm[8], rl[8];
#pragma unroll
  for (int i = 0; i < 8; ++i) { rm[i] = -3.0e38f; rl[i] = 0.f; }

  const float scale = 0.03125f;  // 1/sqrt(EMBED)
  const int qg = q0 + wave * 16;
  const int NT = SEQ / 64;

  issue_tile(0, 0);

  for (int t = 0; t < NT; ++t) {
    const int kv0 = t * 64;
    const int buf = t & 1;
    const _Float16* kbuf = &sK[buf][0];
    const _Float16* vbuf = &sV[buf][0];

    wait_async_all();    // this wave's copies into `buf` are in LDS
    __syncthreads();     // everyone's are; and buf^1 readers (iter t-1) done
    if (t + 1 < NT) issue_tile(kv0 + 64, buf ^ 1);

    // S = Q * K^T : 16(q) x 64(kv) per wave
    v8f s[4];
#pragma unroll
    for (int nt = 0; nt < 4; ++nt) {
      v8f c = {};
#pragma unroll
      for (int ks = 0; ks < 2; ++ks) {
        v16h bf = *(const v16h*)&kbuf[(nt * 16 + ln) * 64 + ks * 32 + hi * 16];
        c = wmma_f16(qa[ks], bf, c);
      }
      s[nt] = c;
    }

    // mask (pre-scale in ref; equivalent huge-negative post-scale) + scale
#pragma unroll
    for (int nt = 0; nt < 4; ++nt) {
      const int kidx = kv0 + nt * 16 + ln;
#pragma unroll
      for (int i = 0; i < 8; ++i) {
        const int qr = qg + i + hi * 8;
        float e = s[nt][i] * scale;
        if (mb[(size_t)qr * SEQ + kidx] == 0) e = -3.0e18f;
        s[nt][i] = e;
      }
    }

    // online softmax; each row's 16 values sit across one 16-lane half
#pragma unroll
    for (int i = 0; i < 8; ++i) {
      float mx = fmaxf(fmaxf(s[0][i], s[1][i]), fmaxf(s[2][i], s[3][i]));
#pragma unroll
      for (int d = 8; d >= 1; d >>= 1) mx = fmaxf(mx, __shfl_xor(mx, d, 32));
      const float nm = fmaxf(rm[i], mx);
      const float sf = __expf(rm[i] - nm);
      rm[i] = nm;
      float ps = 0.f;
#pragma unroll
      for (int nt = 0; nt < 4; ++nt) {
        const float p = __expf(s[nt][i] - nm);
        s[nt][i] = p;
        ps += p;
      }
#pragma unroll
      for (int d = 8; d >= 1; d >>= 1) ps += __shfl_xor(ps, d, 32);
      rl[i] = rl[i] * sf + ps;
#pragma unroll
      for (int nt = 0; nt < 4; ++nt) o[nt][i] *= sf;
    }

    // P: C-layout -> A-layout via per-wave LDS staging
    _Float16* pw = &sP[wave][0];
#pragma unroll
    for (int nt = 0; nt < 4; ++nt)
#pragma unroll
      for (int i = 0; i < 8; ++i)
        pw[(i + hi * 8) * 64 + nt * 16 + ln] = (_Float16)s[nt][i];
#if defined(__gfx1250__)
    asm volatile("s_wait_dscnt 0" ::: "memory");
#else
    __threadfence_block();
#endif

    // O += P * V  (B fragments from V^T rows: contiguous)
#pragma unroll
    for (int nt = 0; nt < 4; ++nt) {
#pragma unroll
      for (int ks = 0; ks < 2; ++ks) {
        v16h pa = load_a_f16(pw + ln * 64, ks * 32, hi);
        v16h vb = *(const v16h*)&vbuf[(nt * 16 + ln) * 64 + ks * 32 + hi * 16];
        o[nt] = wmma_f16(pa, vb, o[nt]);
      }
    }
  }

  // normalize + store context in [B,S,EMBED] f16 (heads concatenated)
#pragma unroll
  for (int nt = 0; nt < 4; ++nt) {
    const int d = nt * 16 + ln;
#pragma unroll
    for (int i = 0; i < 8; ++i) {
      const int qr = qg + i + hi * 8;
      ctx[((size_t)b * SEQ + qr) * EMBED + h * HD + d] =
          (_Float16)(o[nt][i] / rl[i]);
    }
  }
}

// -------------------------------------------------------------------------
// Output projection: out[m,e] = sum_k ctx[m,k] * Wo[e,k] + bo[e]
// One wave computes a 16x64 tile; K loop of 32.
__global__ __launch_bounds__(128) void out_proj(const _Float16* __restrict__ ctx,
                                                const _Float16* __restrict__ Wo16,
                                                const float* __restrict__ bo,
                                                float* __restrict__ out) {
  const int tid  = threadIdx.x;
  const int wave = tid >> 5, lane = tid & 31;
  const int ln = lane & 15, hi = lane >> 4;
  const int gw = blockIdx.x * 4 + wave;
  const int m0 = (gw >> 4) * 16;
  const int n0 = (gw & 15) * 64;

  v8f acc[4] = {};
  const _Float16* arow = ctx + (size_t)(m0 + ln) * EMBED;
  for (int k = 0; k < EMBED; k += 32) {
    v16h a = load_a_f16(arow, k, hi);
#pragma unroll
    for (int nt = 0; nt < 4; ++nt) {
      v16h bf = *(const v16h*)(Wo16 + (size_t)(n0 + nt * 16 + ln) * EMBED + k + hi * 16);
      acc[nt] = wmma_f16(a, bf, acc[nt]);
    }
  }
#pragma unroll
  for (int nt = 0; nt < 4; ++nt) {
    const int e = n0 + nt * 16 + ln;
    const float bias = bo[e];
#pragma unroll
    for (int i = 0; i < 8; ++i) {
      const int m = m0 + i + hi * 8;
      out[(size_t)m * EMBED + e] = acc[nt][i] + bias;
    }
  }
}

// -------------------------------------------------------------------------
extern "C" void kernel_launch(void* const* d_in, const int* in_sizes, int n_in,
                              void* d_out, int out_size, void* d_ws, size_t ws_size,
                              hipStream_t stream) {
  const float* values  = (const float*)d_in[0];
  const float* keys    = (const float*)d_in[1];
  const float* queries = (const float*)d_in[2];
  const int*   mask    = (const int*)d_in[3];
  const float* Wv = (const float*)d_in[4];
  const float* Wk = (const float*)d_in[5];
  const float* Wq = (const float*)d_in[6];
  const float* Wo = (const float*)d_in[7];
  const float* bo = (const float*)d_in[8];
  float* out = (float*)d_out;

  char* ws = (char*)d_ws;
  const size_t TEN = (size_t)BATCH * HEADS * SEQ * HD * sizeof(_Float16);  // 16 MB
  _Float16* Qh  = (_Float16*)(ws);            // [B,H,S,D]
  _Float16* Kh  = (_Float16*)(ws + TEN);      // [B,H,S,D]
  _Float16* Vt  = (_Float16*)(ws + 2 * TEN);  // [B,H,D,S]
  _Float16* ctx = (_Float16*)(ws + 3 * TEN);  // [B,S,EMBED]
  _Float16* Wo16 = (_Float16*)(ws + 4 * TEN);
  _Float16* Wq16 = (_Float16*)(ws + 4 * TEN + (size_t)EMBED * EMBED * sizeof(_Float16));
  _Float16* Wk16 = Wq16 + HD * HD;
  _Float16* Wv16 = Wk16 + HD * HD;

  cvt_f32_to_f16<<<(EMBED * EMBED + 255) / 256, 256, 0, stream>>>(Wo, Wo16, EMBED * EMBED);
  cvt_f32_to_f16<<<(HD * HD + 255) / 256, 256, 0, stream>>>(Wq, Wq16, HD * HD);
  cvt_f32_to_f16<<<(HD * HD + 255) / 256, 256, 0, stream>>>(Wk, Wk16, HD * HD);
  cvt_f32_to_f16<<<(HD * HD + 255) / 256, 256, 0, stream>>>(Wv, Wv16, HD * HD);

  const int nblk_proj = (BATCH * SEQ * HEADS / 16) / 4;  // 2048 blocks of 4 waves
  proj16<<<nblk_proj, 128, 0, stream>>>(queries, Wq16, Qh, 0);
  proj16<<<nblk_proj, 128, 0, stream>>>(keys,    Wk16, Kh, 0);
  proj16<<<nblk_proj, 128, 0, stream>>>(values,  Wv16, Vt, 1);

  dim3 ag(SEQ / 64, HEADS, BATCH);
  flash_attn<<<ag, 128, 0, stream>>>(Qh, Kh, Vt, mask, ctx);

  const int nblk_out = ((BATCH * SEQ / 16) * (EMBED / 64)) / 4;  // 2048
  out_proj<<<nblk_out, 128, 0, stream>>>(ctx, Wo16, bo, out);

  (void)in_sizes; (void)n_in; (void)out_size; (void)ws_size;
}
